// MultiScaleAttention_31404800868646
// MI455X (gfx1250) — matrix operand order repaired and checked by reading, not compiled, not run on hardware
//
#include <hip/hip_runtime.h>

typedef __attribute__((ext_vector_type(16))) __bf16 v16bf;
typedef __attribute__((ext_vector_type(8)))  float  v8f;
typedef __attribute__((ext_vector_type(4)))  int    v4i_t;

#define B_   8
#define NTOK 16384     // 128*128
#define HEADS 4
#define CHD  64
#define SCALE_F 0.125f

#if defined(__has_builtin)
#if __has_builtin(__builtin_amdgcn_global_load_async_to_lds_b128)
#define HAVE_ASYNC_B128 1
#endif
#endif

#define AS1_ __attribute__((address_space(1)))
#define AS3_ __attribute__((address_space(3)))

__device__ inline void copy16_async(const void* g, void* l) {
#ifdef HAVE_ASYNC_B128
  __builtin_amdgcn_global_load_async_to_lds_b128(
      (AS1_ v4i_t*)g, (AS3_ v4i_t*)l, 0, 0);
#else
  *(uint4*)l = *(const uint4*)g;
#endif
}

__device__ inline void async_wait_all() {
#ifdef HAVE_ASYNC_B128
  asm volatile("s_wait_asynccnt 0" ::: "memory");
#endif
}

__device__ inline v8f zero8() {
  v8f v;
#pragma unroll
  for (int i = 0; i < 8; ++i) v[i] = 0.0f;
  return v;
}

__device__ inline v8f wmma_bf16(v16bf a, v16bf b, v8f c) {
  return __builtin_amdgcn_wmma_f32_16x16x32_bf16(false, a, false, b, (short)0, c,
                                                 false, false);
}

// A-fragment (16x32, bf16) from row-major LDS tile: element (m,k) at sm[m*ldm+k].
// Lane L<16: row L, K = {kb..kb+7, kb+16..kb+23}, kb = k0 (+8 for lanes >= 16).
// Lowers to 2x ds_load_b128 per lane.
__device__ inline v16bf lda16(const __bf16* sm, int ldm, int row0, int k0, int lane) {
  int m  = row0 + (lane & 15);
  int kb = k0 + ((lane & 16) >> 1);
  const __bf16* p = sm + m * ldm + kb;
  v16bf f;
#pragma unroll
  for (int h = 0; h < 8; ++h) { f[h] = p[h]; f[h + 8] = p[h + 16]; }
  return f;
}

// B-fragment (32x16, bf16) from an N-MAJOR LDS tile: element (k,n) at sm[n*ldn+k].
// Lane L<16: col n=n0+L, K = {kb..kb+7, kb+16..kb+23}, kb = k0 (+8 for lanes >= 16).
// 2x 8 consecutive bf16 per lane -> 2x ds_load_b128.
__device__ inline v16bf ldbT16(const __bf16* sm, int ldn, int k0, int n0, int lane) {
  int n  = n0 + (lane & 15);
  int kb = k0 + ((lane & 16) >> 1);
  const __bf16* p = sm + n * ldn + kb;
  v16bf f;
#pragma unroll
  for (int h = 0; h < 8; ++h) { f[h] = p[h]; f[h + 8] = p[h + 16]; }
  return f;
}

// ---------------------------------------------------------------------------
// Fused QKV projection: X(131072x256) @ [Wq | Wkv](256x768) -> Q,K,V bf16
// layout (B, HEADS, NTOK, 64). 128x128 tile per block, 8 waves in 4x2 grid.
// ---------------------------------------------------------------------------
__global__ __launch_bounds__(256)
void qkv_gemm(const float* __restrict__ X, const float* __restrict__ Wq,
              const float* __restrict__ Wkv,
              __bf16* __restrict__ Qo, __bf16* __restrict__ Ko,
              __bf16* __restrict__ Vo) {
  __shared__ __bf16 a_sm[128][40];    // [row][k]
  __shared__ __bf16 bT_sm[128][40];   // [col][k]  (N-major for B-fragments)
  const int tid  = (int)threadIdx.x;
  const int lane = tid & 31;
  const int wave = tid >> 5;
  const int rowBase = (int)blockIdx.x * 128;
  const int colBase = (int)blockIdx.y * 128;
  const int wm = (wave >> 1) * 32;
  const int wn = (wave & 1) * 64;

  v8f acc[2][4];
#pragma unroll
  for (int mi = 0; mi < 2; ++mi)
#pragma unroll
    for (int ni = 0; ni < 4; ++ni) acc[mi][ni] = zero8();

  for (int kt = 0; kt < 8; ++kt) {
    __syncthreads();
    { // stage A (128x32 f32 -> bf16)
      int r = tid >> 1, seg = (tid & 1) * 16;
      const float* src = X + (size_t)(rowBase + r) * 256 + kt * 32 + seg;
      if (kt < 7) __builtin_prefetch(src + 32, 0, 1);
#pragma unroll
      for (int c = 0; c < 16; ++c) a_sm[r][seg + c] = (__bf16)src[c];
    }
    { // stage B transposed: coalesced row reads of W, N-major LDS writes
      int kr = tid >> 3, seg = (tid & 7) * 16;
      int krow = kt * 32 + kr;
#pragma unroll
      for (int c = 0; c < 16; ++c) {
        int col = colBase + seg + c;
        float w = (col < 256) ? Wq[(size_t)krow * 256 + col]
                              : Wkv[(size_t)krow * 512 + (col - 256)];
        bT_sm[seg + c][kr] = (__bf16)w;
      }
    }
    __syncthreads();
    v16bf afr[2];
#pragma unroll
    for (int mi = 0; mi < 2; ++mi)
      afr[mi] = lda16(&a_sm[0][0], 40, wm + mi * 16, 0, lane);
#pragma unroll
    for (int ni = 0; ni < 4; ++ni) {
      v16bf bfr = ldbT16(&bT_sm[0][0], 40, 0, wn + ni * 16, lane);
#pragma unroll
      for (int mi = 0; mi < 2; ++mi)
        acc[mi][ni] = wmma_bf16(afr[mi], bfr, acc[mi][ni]);
    }
  }

  // epilogue: scatter into per-head Q/K/V bf16
#pragma unroll
  for (int mi = 0; mi < 2; ++mi)
#pragma unroll
    for (int ni = 0; ni < 4; ++ni)
#pragma unroll
      for (int r = 0; r < 8; ++r) {
        int row = rowBase + wm + mi * 16 + ((lane & 16) >> 1) + r;
        int col = colBase + wn + ni * 16 + (lane & 15);
        int bb = row >> 14, n = row & 16383;
        __bf16 bv = (__bf16)acc[mi][ni][r];
        if (col < 256) {
          Qo[(((size_t)bb * HEADS + (col >> 6)) * NTOK + n) * CHD + (col & 63)] = bv;
        } else {
          int j = col - 256;
          __bf16* dst = (j >> 8) ? Vo : Ko;
          int h = (j >> 6) & 3;
          dst[(((size_t)bb * HEADS + h) * NTOK + n) * CHD + (j & 63)] = bv;
        }
      }
}

// ---------------------------------------------------------------------------
// Patch-order token -> raster index for a PSIDE x PSIDE windowed head.
// ---------------------------------------------------------------------------
template<int PSIDE>
__device__ inline int patch2raster(int tile, int j) {
  constexpr int WIN = PSIDE * PSIDE;
  constexpr int WPL = 128 / PSIDE;
  int tpi = tile * 256 + j;
  int win = tpi / WIN, pos = tpi % WIN;
  return ((win / WPL) * PSIDE + pos / PSIDE) * 128 + (win % WPL) * PSIDE + (pos % PSIDE);
}

// ---------------------------------------------------------------------------
// Flash-style masked window attention over a 256-token patch-ordered tile.
// GLOBALQ=true: Q = gq (256x64 per batch, broadcast over windows), no mask,
// accumulate-add result into `fused`.
// ---------------------------------------------------------------------------
#define LDQ_  72
#define LDK_  72
#define LDVT_ 40
#define LDP_  40
#define ATTN_SMEM_BYTES ((256 * LDQ_ + 32 * LDK_ + 64 * LDVT_ + 8 * 32 * LDP_) * 2)

template<int PSIDE, bool GLOBALQ>
__global__ __launch_bounds__(256)
void attn_kernel(const __bf16* __restrict__ Qg, const __bf16* __restrict__ Kg,
                 const __bf16* __restrict__ Vg, __bf16* __restrict__ fused,
                 int head) {
  constexpr int WIN = PSIDE * PSIDE;
  extern __shared__ char smem_raw[];
  __bf16* q_sm  = (__bf16*)smem_raw;           // 256 x LDQ_   [row][ch]
  __bf16* k_sm  = q_sm + 256 * LDQ_;           // 32 x LDK_    [key][ch]
  __bf16* vT_sm = k_sm + 32 * LDK_;            // 64 x LDVT_   [ch][key] transposed
  __bf16* p_sm  = vT_sm + 64 * LDVT_;          // 8 waves x 32 x LDP_

  const int tid  = (int)threadIdx.x;
  const int lane = tid & 31;
  const int wave = tid >> 5;
  const int b    = (int)blockIdx.x >> 6;
  const int tile = (int)blockIdx.x & 63;
  const size_t kvbase = ((size_t)b * HEADS + head) * NTOK;

  { // stage Q: one 64-ch row per thread via async global->LDS (16B granules)
    int row = tid;
    const __bf16* src;
    if (GLOBALQ) {
      src = Qg + ((size_t)b * 256 + row) * CHD;
    } else {
      int n = patch2raster<PSIDE>(tile, row);
      src = Qg + (kvbase + n) * CHD;
    }
#pragma unroll
    for (int c = 0; c < CHD; c += 8)
      copy16_async(src + c, &q_sm[row * LDQ_ + c]);
  }

  float run_m[2][8], run_l[2][8];
#pragma unroll
  for (int mi = 0; mi < 2; ++mi)
#pragma unroll
    for (int r = 0; r < 8; ++r) { run_m[mi][r] = -1e29f; run_l[mi][r] = 0.0f; }

  v8f oacc[2][4];
#pragma unroll
  for (int mi = 0; mi < 2; ++mi)
#pragma unroll
    for (int ci = 0; ci < 4; ++ci) oacc[mi][ci] = zero8();

  const int m0 = wave * 32;
  __bf16* p_wave = p_sm + wave * 32 * LDP_;
  const int rHi = (lane & 16) >> 1;
  const int nLo = lane & 15;

  for (int kc = 0; kc < 8; ++kc) {
    __syncthreads();
    { // stage K (async, row-major) and V (transposed scalar stores)
      int r = tid >> 3, seg = (tid & 7) * 8;
      int n = patch2raster<PSIDE>(tile, kc * 32 + r);
      const __bf16* ks = Kg + (kvbase + n) * CHD + seg;
      const __bf16* vs = Vg + (kvbase + n) * CHD + seg;
      copy16_async(ks, &k_sm[r * LDK_ + seg]);
#pragma unroll
      for (int c = 0; c < 8; ++c) vT_sm[(seg + c) * LDVT_ + r] = vs[c];
      if (kc < 7) {
        int n2 = patch2raster<PSIDE>(tile, kc * 32 + 32 + r);
        __builtin_prefetch(Kg + (kvbase + n2) * CHD, 0, 1);
        __builtin_prefetch(Vg + (kvbase + n2) * CHD, 0, 1);
      }
    }
    async_wait_all();   // covers Q staging on first iteration as well
    __syncthreads();

    // S = Q Kt (32 rows x 32 keys per wave), K-dim = 64 channels
    v8f s[2][2];
#pragma unroll
    for (int mi = 0; mi < 2; ++mi)
#pragma unroll
      for (int ni = 0; ni < 2; ++ni) {
        v8f a = zero8();
#pragma unroll
        for (int kk = 0; kk < CHD; kk += 32) {
          v16bf af = lda16(q_sm, LDQ_, m0 + mi * 16, kk, lane);
          v16bf bf = ldbT16(k_sm, LDK_, kk, ni * 16, lane);  // (ch, key), contig
          a = wmma_bf16(af, bf, a);
        }
        s[mi][ni] = a;
      }

    // online softmax per row (16-lane reductions within half-wave groups)
#pragma unroll
    for (int mi = 0; mi < 2; ++mi) {
      float pv[2][8], mx8[8];
#pragma unroll
      for (int r = 0; r < 8; ++r) {
        int row = m0 + mi * 16 + rHi + r;
        float mx = -1e30f;
#pragma unroll
        for (int ni = 0; ni < 2; ++ni) {
          float v = s[mi][ni][r] * SCALE_F;
          if (!GLOBALQ) {
            int col = kc * 32 + ni * 16 + nLo;
            if ((row & ~(WIN - 1)) != (col & ~(WIN - 1))) v = -1e30f;
          }
          pv[ni][r] = v;
          mx = fmaxf(mx, v);
        }
        mx8[r] = mx;
      }
#pragma unroll
      for (int r = 0; r < 8; ++r)
#pragma unroll
        for (int off = 1; off < 16; off <<= 1)
          mx8[r] = fmaxf(mx8[r], __shfl_xor(mx8[r], off, 32));

#pragma unroll
      for (int r = 0; r < 8; ++r) {
        float mnew = fmaxf(fmaxf(run_m[mi][r], mx8[r]), -1e29f);
        float alpha = __expf(run_m[mi][r] - mnew);
        run_m[mi][r] = mnew;
        float ssum = 0.0f;
#pragma unroll
        for (int ni = 0; ni < 2; ++ni) {
          float e = __expf(pv[ni][r] - mnew);
          pv[ni][r] = e;
          ssum += e;
        }
#pragma unroll
        for (int off = 1; off < 16; off <<= 1) ssum += __shfl_xor(ssum, off, 32);
        run_l[mi][r] = run_l[mi][r] * alpha + ssum;
#pragma unroll
        for (int ci = 0; ci < 4; ++ci) oacc[mi][ci][r] *= alpha;
#pragma unroll
        for (int ni = 0; ni < 2; ++ni)
          p_wave[(mi * 16 + rHi + r) * LDP_ + ni * 16 + nLo] = (__bf16)pv[ni][r];
      }
    }
    __syncthreads();

    // O += P(32x32) @ V(32x64); V B-fragments contiguous via vT_sm
#pragma unroll
    for (int mi = 0; mi < 2; ++mi) {
      v16bf af = lda16(p_wave, LDP_, mi * 16, 0, lane);
#pragma unroll
      for (int ci = 0; ci < 4; ++ci) {
        v16bf bf = ldbT16(vT_sm, LDVT_, 0, ci * 16, lane);   // (key, ch)
        oacc[mi][ci] = wmma_bf16(af, bf, oacc[mi][ci]);
      }
    }
  }

  // normalize and write (patch-order token index, head slice of fused)
#pragma unroll
  for (int mi = 0; mi < 2; ++mi)
#pragma unroll
    for (int ci = 0; ci < 4; ++ci)
#pragma unroll
      for (int r = 0; r < 8; ++r) {
        float o = oacc[mi][ci][r] / run_l[mi][r];
        int row = m0 + mi * 16 + rHi + r;
        int col = ci * 16 + nLo;
        size_t dst = ((size_t)b * NTOK + tile * 256 + row) * 256 + head * CHD + col;
        if (GLOBALQ) o += (float)fused[dst];
        fused[dst] = (__bf16)o;
      }
}

// ---------------------------------------------------------------------------
// gs[b][c][py][px] = mean over windows of head output at within-window pos.
// ---------------------------------------------------------------------------
__global__ __launch_bounds__(256)
void gs_mean_kernel(const __bf16* __restrict__ fused, float* __restrict__ gs,
                    int head, int P) {
  int PP = P * P;
  int total = B_ * CHD * PP;
  int idx = (int)blockIdx.x * 256 + (int)threadIdx.x;
  if (idx >= total) return;
  int b = idx / (CHD * PP);
  int rem = idx % (CHD * PP);
  int c = rem / PP, pp = rem % PP;
  int NP = NTOK / PP;
  float sum = 0.0f;
  for (int w = 0; w < NP; ++w) {
    int token = w * PP + pp;
    sum += (float)fused[((size_t)b * NTOK + token) * 256 + head * CHD + c];
  }
  gs[(((size_t)b * CHD + c) * P + pp / P) * P + (pp % P)] = sum / (float)NP;
}

// cur[b][c][2s][2s] += bilinear_upsample(prev[b][c][s][s])  (half-pixel centers)
__global__ __launch_bounds__(256)
void up_add_kernel(const float* __restrict__ prev, float* __restrict__ cur, int s) {
  int S = s * 2;
  int total = B_ * CHD * S * S;
  int idx = (int)blockIdx.x * 256 + (int)threadIdx.x;
  if (idx >= total) return;
  int bc = idx / (S * S);
  int yx = idx % (S * S);
  int y = yx / S, x = yx % S;
  float fy = (y + 0.5f) * 0.5f - 0.5f;
  float fx = (x + 0.5f) * 0.5f - 0.5f;
  int y0 = (int)floorf(fy), x0 = (int)floorf(fx);
  float wy = fy - (float)y0, wx = fx - (float)x0;
  int y1 = min(max(y0 + 1, 0), s - 1), x1 = min(max(x0 + 1, 0), s - 1);
  y0 = min(max(y0, 0), s - 1);  x0 = min(max(x0, 0), s - 1);
  const float* p = prev + (size_t)bc * s * s;
  float v = (1.0f - wy) * ((1.0f - wx) * p[y0 * s + x0] + wx * p[y0 * s + x1]) +
            wy          * ((1.0f - wx) * p[y1 * s + x0] + wx * p[y1 * s + x1]);
  cur[idx] += v;
}

// gq[b][t][c] = gl4.reshape(B,-1,64) scramble of gl4 (B,64,16,16), f32 -> bf16
__global__ __launch_bounds__(256)
void gq_pack_kernel(const float* __restrict__ gl, __bf16* __restrict__ gq) {
  int total = B_ * 256 * CHD;
  int idx = (int)blockIdx.x * 256 + (int)threadIdx.x;
  if (idx >= total) return;
  int b = idx / (256 * CHD);
  int t2 = idx % (256 * CHD);           // t*64 + c == flat over (64,16,16)
  int ch = t2 >> 8, rem = t2 & 255;
  int y = rem >> 4, x = rem & 15;
  gq[(size_t)b * 256 * CHD + t2] =
      (__bf16)gl[(((size_t)b * CHD + ch) * 16 + y) * 16 + x];
}

// ---------------------------------------------------------------------------
// Output projection: fused(131072x256 bf16) @ Wproj(256x256) + bproj -> f32
// ---------------------------------------------------------------------------
__global__ __launch_bounds__(256)
void proj_gemm(const __bf16* __restrict__ A, const float* __restrict__ Wp,
               const float* __restrict__ bias, float* __restrict__ out) {
  __shared__ __bf16 a_sm[128][40];
  __shared__ __bf16 bT_sm[128][40];   // [col][k]
  const int tid  = (int)threadIdx.x;
  const int lane = tid & 31;
  const int wave = tid >> 5;
  const int rowBase = (int)blockIdx.x * 128;
  const int colBase = (int)blockIdx.y * 128;
  const int wm = (wave >> 1) * 32;
  const int wn = (wave & 1) * 64;

  v8f acc[2][4];
#pragma unroll
  for (int mi = 0; mi < 2; ++mi)
#pragma unroll
    for (int ni = 0; ni < 4; ++ni) acc[mi][ni] = zero8();

  for (int kt = 0; kt < 8; ++kt) {
    __syncthreads();
    {
      int r = tid >> 1, seg = (tid & 1) * 16;
      const __bf16* src = A + (size_t)(rowBase + r) * 256 + kt * 32 + seg;
      if (kt < 7) __builtin_prefetch(src + 32, 0, 1);
#pragma unroll
      for (int c = 0; c < 16; ++c) a_sm[r][seg + c] = src[c];
    }
    {
      int kr = tid >> 3, seg = (tid & 7) * 16;
      int krow = kt * 32 + kr;
#pragma unroll
      for (int c = 0; c < 16; ++c)
        bT_sm[seg + c][kr] = (__bf16)Wp[(size_t)krow * 256 + colBase + seg + c];
    }
    __syncthreads();
    v16bf afr[2];
#pragma unroll
    for (int mi = 0; mi < 2; ++mi)
      afr[mi] = lda16(&a_sm[0][0], 40, wm + mi * 16, 0, lane);
#pragma unroll
    for (int ni = 0; ni < 4; ++ni) {
      v16bf bfr = ldbT16(&bT_sm[0][0], 40, 0, wn + ni * 16, lane);
#pragma unroll
      for (int mi = 0; mi < 2; ++mi)
        acc[mi][ni] = wmma_bf16(afr[mi], bfr, acc[mi][ni]);
    }
  }

#pragma unroll
  for (int mi = 0; mi < 2; ++mi)
#pragma unroll
    for (int ni = 0; ni < 4; ++ni)
#pragma unroll
      for (int r = 0; r < 8; ++r) {
        int row = rowBase + wm + mi * 16 + ((lane & 16) >> 1) + r;
        int col = colBase + wn + ni * 16 + (lane & 15);
        out[(size_t)row * 256 + col] = acc[mi][ni][r] + bias[col];
      }
}

// ---------------------------------------------------------------------------
extern "C" void kernel_launch(void* const* d_in, const int* in_sizes, int n_in,
                              void* d_out, int out_size, void* d_ws, size_t ws_size,
                              hipStream_t stream) {
  (void)in_sizes; (void)n_in; (void)out_size; (void)ws_size;
  const float* x     = (const float*)d_in[0];
  const float* Wq    = (const float*)d_in[1];
  const float* Wkv   = (const float*)d_in[2];
  const float* Wproj = (const float*)d_in[3];
  const float* bproj = (const float*)d_in[4];
  float* out = (float*)d_out;

  char* ws = (char*)d_ws;
  size_t off = 0;
  auto carve = [&](size_t bytes) {
    char* p = ws + off;
    off += (bytes + 255) & ~(size_t)255;
    return p;
  };
  const size_t qkv_bytes = (size_t)B_ * HEADS * NTOK * CHD * sizeof(__bf16);  // 64 MiB
  __bf16* Q     = (__bf16*)carve(qkv_bytes);
  __bf16* K     = (__bf16*)carve(qkv_bytes);
  __bf16* V     = (__bf16*)carve(qkv_bytes);
  __bf16* fused = (__bf16*)carve((size_t)B_ * NTOK * 256 * sizeof(__bf16));
  float*  g2    = (float*)carve((size_t)B_ * CHD * 2 * 2 * sizeof(float));
  float*  g4    = (float*)carve((size_t)B_ * CHD * 4 * 4 * sizeof(float));
  float*  g8    = (float*)carve((size_t)B_ * CHD * 8 * 8 * sizeof(float));
  float*  g16   = (float*)carve((size_t)B_ * CHD * 16 * 16 * sizeof(float));
  __bf16* gq    = (__bf16*)carve((size_t)B_ * 256 * CHD * sizeof(__bf16));

  // 1) fused QKV projection
  qkv_gemm<<<dim3(1024, 6), 256, 0, stream>>>(x, Wq, Wkv, Q, K, V);

  // 2) window attention per head (512 tiles = B * 64 per head)
  attn_kernel<2,  false><<<512, 256, ATTN_SMEM_BYTES, stream>>>(Q, K, V, fused, 0);
  attn_kernel<4,  false><<<512, 256, ATTN_SMEM_BYTES, stream>>>(Q, K, V, fused, 1);
  attn_kernel<8,  false><<<512, 256, ATTN_SMEM_BYTES, stream>>>(Q, K, V, fused, 2);
  attn_kernel<16, false><<<512, 256, ATTN_SMEM_BYTES, stream>>>(Q, K, V, fused, 3);

  // 3) hierarchical global feature: per-head window means + bilinear chain
  gs_mean_kernel<<<(B_ * CHD * 4   + 255) / 256, 256, 0, stream>>>(fused, g2, 0, 2);
  gs_mean_kernel<<<(B_ * CHD * 16  + 255) / 256, 256, 0, stream>>>(fused, g4, 1, 4);
  gs_mean_kernel<<<(B_ * CHD * 64  + 255) / 256, 256, 0, stream>>>(fused, g8, 2, 8);
  gs_mean_kernel<<<(B_ * CHD * 256 + 255) / 256, 256, 0, stream>>>(fused, g16, 3, 16);
  up_add_kernel<<<(B_ * CHD * 16  + 255) / 256, 256, 0, stream>>>(g2, g4, 2);
  up_add_kernel<<<(B_ * CHD * 64  + 255) / 256, 256, 0, stream>>>(g4, g8, 4);
  up_add_kernel<<<(B_ * CHD * 256 + 255) / 256, 256, 0, stream>>>(g8, g16, 8);
  gq_pack_kernel<<<(B_ * 256 * CHD + 255) / 256, 256, 0, stream>>>(g16, gq);

  // 4) broadcast global attention on head 3, accumulated into fused
  attn_kernel<16, true><<<512, 256, ATTN_SMEM_BYTES, stream>>>(gq, K, V, fused, 3);

  // 5) output projection
  proj_gemm<<<dim3(1024, 2), 256, 0, stream>>>(fused, Wproj, bproj, out);
}
